// CAGroup3DHead_23922967838982
// MI455X (gfx1250) — matrix operand order repaired and checked by reading, not compiled, not run on hardware
//
#include <hip/hip_runtime.h>
#include <cstddef>

// ---------------------------------------------------------------------------
// CAGroup3DHead forward for MI455X (gfx1250, wave32, WMMA + TDM).
// Heavy GEMMs go through v_wmma_f32_16x16x32_f16 (f16 in, f32 accumulate).
// Per-class weights are prefetched with the Tensor Data Mover (async DMA to
// LDS, double-buffered, s_wait_tensorcnt) so the DMA overlaps WMMA compute.
// ---------------------------------------------------------------------------

#define N_PTS 131072
#define C_FEAT 64
#define N_CLS 18
#define VOXEL 0.04f
// sigmoid(x) > 0.15  <=>  x > log(0.15/0.85)
#define SEM_LOGIT -1.7346011f

typedef __attribute__((ext_vector_type(16))) _Float16 v16h;
typedef __attribute__((ext_vector_type(8)))  float    v8f;
typedef __attribute__((ext_vector_type(4)))  unsigned u32x4;
typedef __attribute__((ext_vector_type(8)))  int      i32x8;
typedef __attribute__((ext_vector_type(4)))  int      i32x4;

#define A_STRIDE 68   // f16 elems per A-tile row in LDS (padded vs 64)
#define B_STRIDE 66   // f16 elems per B^T row in LDS (padded vs 64)

#if __has_builtin(__builtin_amdgcn_tensor_load_to_lds) && \
    __has_builtin(__builtin_amdgcn_s_wait_tensorcnt)
#define HAVE_TDM 1
#else
#define HAVE_TDM 0
#endif

__device__ __forceinline__ v8f wmma16(v16h a, v16h b, v8f c) {
  // 8 args: (neg_a, A, neg_b, B, c_mod, C, reuse_a, reuse_b)
  return __builtin_amdgcn_wmma_f32_16x16x32_f16(false, a, false, b, (short)0, c,
                                                false, false);
}

#if HAVE_TDM
// Async DMA of a 64x64 fp32 tensor (row-major, contiguous) into LDS via the
// Tensor Data Mover. D# layout per CDNA5 ISA:
//   group0: count=1 | lds_addr | global_addr[56:0] | type=2
//   group1: data_size=4B, tensor_dim0/1=64, tile_dim0/1=64, dim0_stride=64,
//           workgroup_mask=0 (not in a cluster), no padding/iterate/barrier.
__device__ __forceinline__ void tdm_load_w64x64(const float* gsrc, void* lds_dst) {
  const unsigned lds_off = (unsigned)(unsigned long long)lds_dst; // aperture low bits = LDS offset
  const unsigned long long ga = (unsigned long long)gsrc;
  u32x4 g0 = { 1u,                                      // count=1 (valid descriptor)
               lds_off,                                 // lds_addr  (bits 63:32)
               (unsigned)(ga & 0xffffffffu),            // global_addr lo
               (unsigned)((ga >> 32) & 0x01ffffffu) |   // global_addr[56:32]
                   (2u << 30) };                        // type=2 ("image")
  i32x8 g1 = { (int)(2u << 16),    // wg_mask=0, data_size=2 (4 bytes)
               (int)(64u << 16),   // tensor_dim0 = 64 (low half in [31:16])
               (int)(64u << 16),   // tensor_dim0 hi=0 | tensor_dim1 = 64
               (int)(64u << 16),   // tensor_dim1 hi=0 | tile_dim0 = 64
               64,                 // tile_dim1 = 64, tile_dim2 = 0
               64,                 // tensor_dim0_stride = 64
               0, 0 };             // dim0_stride hi, dim1_stride = 0
  i32x4 z4 = {0, 0, 0, 0};
#if __clang_major__ >= 23
  i32x8 z8 = {0, 0, 0, 0, 0, 0, 0, 0};
  __builtin_amdgcn_tensor_load_to_lds(g0, g1, z4, z4, z8, 0);
#else
  __builtin_amdgcn_tensor_load_to_lds(g0, g1, z4, z4, 0);
#endif
}
#endif

// Gather the 16x32 f16 A operand for k-step `ks` from an LDS tile a[16][A_STRIDE].
// ISA layout: lanes 0-15 hold M=lane, K = {0..7, 16..23}; lanes 16-31 hold
// M=lane-16, K = {8..15, 24..31}; pairs packed per VGPR.
__device__ __forceinline__ v16h gather_a(const _Float16* a, int lane, int ks) {
  const int row = lane & 15;
  const int kb  = ks * 32 + ((lane >> 4) << 3);
  const _Float16* p = a + row * A_STRIDE + kb;
  v16h r;
#pragma unroll
  for (int j = 0; j < 4; ++j) { r[2 * j] = p[2 * j]; r[2 * j + 1] = p[2 * j + 1]; }
#pragma unroll
  for (int j = 0; j < 4; ++j) { r[8 + 2 * j] = p[16 + 2 * j]; r[9 + 2 * j] = p[17 + 2 * j]; }
  return r;
}

// Gather the 32x16 f16 B operand (k-step ks, 16-col tile nt) from an LDS
// buffer holding W^T: bt[n][k] with row stride B_STRIDE. ISA layout: lanes
// 0-15 hold K = {0..7, 16..23}, lanes 16-31 hold K = {8..15, 24..31}, N = lane&15.
__device__ __forceinline__ v16h gather_b(const _Float16* bt, int lane, int ks, int nt) {
  const int n  = nt * 16 + (lane & 15);
  const int kb = ks * 32 + ((lane >> 4) << 3);
  const _Float16* p = bt + n * B_STRIDE + kb;
  v16h r;
#pragma unroll
  for (int j = 0; j < 8; ++j) r[j] = p[j];
#pragma unroll
  for (int j = 0; j < 8; ++j) r[8 + j] = p[16 + j];
  return r;
}

// Coalesced load of a 16x64 fp32 tile -> f16 LDS tile (per-wave, 256 float4s).
__device__ __forceinline__ void load_a_tile(const float* __restrict__ feats_row0,
                                            _Float16* a, int lane) {
  const float4* src = (const float4*)feats_row0;
#pragma unroll
  for (int it = 0; it < 8; ++it) {
    const int f = it * 32 + lane;           // consecutive lanes -> consecutive float4
    float4 v = src[f];
    const int row = f >> 4, col = (f & 15) << 2;
    _Float16* dst = a + row * A_STRIDE + col;
    dst[0] = (_Float16)v.x; dst[1] = (_Float16)v.y;
    dst[2] = (_Float16)v.z; dst[3] = (_Float16)v.w;
  }
}

__device__ __forceinline__ float elu1(float v) {
  return v > 0.0f ? v : (__expf(v) - 1.0f);
}

// ---------------------------------------------------------------------------
// Kernel 1: scene bounds (int min/max of voxel coords)
// ---------------------------------------------------------------------------
__global__ void minmax_init(int* mm) {
  const int t = threadIdx.x;
  if (t < 8) mm[t] = (t < 4) ? 0x7fffffff : (int)0x80000000;
}

__global__ __launch_bounds__(256) void minmax_kernel(const int* __restrict__ coords,
                                                     int* __restrict__ mm) {
  __shared__ int s[8];
  const int tid = threadIdx.x;
  if (tid < 8) s[tid] = (tid < 4) ? 0x7fffffff : (int)0x80000000;
  __syncthreads();
  int lmin[3] = {0x7fffffff, 0x7fffffff, 0x7fffffff};
  int lmax[3] = {(int)0x80000000, (int)0x80000000, (int)0x80000000};
  const int4* c4 = (const int4*)coords;
  for (int i = blockIdx.x * 256 + tid; i < N_PTS; i += gridDim.x * 256) {
    int4 v = c4[i];
    lmin[0] = min(lmin[0], v.y); lmax[0] = max(lmax[0], v.y);
    lmin[1] = min(lmin[1], v.z); lmax[1] = max(lmax[1], v.z);
    lmin[2] = min(lmin[2], v.w); lmax[2] = max(lmax[2], v.w);
  }
#pragma unroll
  for (int k = 0; k < 3; ++k) { atomicMin(&s[k], lmin[k]); atomicMax(&s[4 + k], lmax[k]); }
  __syncthreads();
  if (tid < 3) atomicMin(&mm[tid], s[tid]);
  if (tid >= 4 && tid < 7) atomicMax(&mm[tid], s[tid]);
}

// ---------------------------------------------------------------------------
// Kernel 2: semantic scores -> per-point class mask, offset MLP -> voted xyz.
// One 16-row tile per wave; WMMA for all four GEMMs.
// ---------------------------------------------------------------------------
__global__ __launch_bounds__(256) void offset_sem_vote_kernel(
    const int* __restrict__ coords, const float* __restrict__ feats,
    const float* __restrict__ W_sem, const float* __restrict__ b_sem,
    const float* __restrict__ W_o1, const float* __restrict__ g_o1,
    const float* __restrict__ b_o1, const float* __restrict__ W_o2,
    const float* __restrict__ g_o2, const float* __restrict__ b_o2,
    const float* __restrict__ W_o3, const int* __restrict__ mm,
    unsigned* __restrict__ maskw, float* __restrict__ voted) {
  __shared__ _Float16 W1t[64 * B_STRIDE];   // W_o1^T
  __shared__ _Float16 W2t[64 * B_STRIDE];   // W_o2^T
  __shared__ _Float16 W3t[16 * B_STRIDE];   // W_o3^T, cols 3..15 zero
  __shared__ _Float16 WsT[32 * B_STRIDE];   // W_sem^T, cols 18..31 zero
  __shared__ _Float16 Ash[8][16 * A_STRIDE];
  __shared__ _Float16 Hsh[8][16 * A_STRIDE];
  __shared__ unsigned mwb[8][16];

  const int tid = threadIdx.x;
  const int wave = tid >> 5, lane = tid & 31;

  for (int i = tid; i < 4096; i += 256) {
    const int d = i >> 6, e = i & 63;
    W1t[e * B_STRIDE + d] = (_Float16)W_o1[i];
    W2t[e * B_STRIDE + d] = (_Float16)W_o2[i];
  }
  for (int i = tid; i < 16 * 64; i += 256) {
    const int j = i >> 6, d = i & 63;
    W3t[j * B_STRIDE + d] = (_Float16)(j < 3 ? W_o3[d * 3 + j] : 0.0f);
  }
  for (int i = tid; i < 32 * 64; i += 256) {
    const int j = i >> 6, d = i & 63;
    WsT[j * B_STRIDE + d] = (_Float16)(j < N_CLS ? W_sem[d * N_CLS + j] : 0.0f);
  }
  __syncthreads();

  const int rows0 = blockIdx.x * 128 + wave * 16;
  _Float16* A = Ash[wave];
  _Float16* H = Hsh[wave];
  load_a_tile(feats + (size_t)rows0 * C_FEAT, A, lane);

  const v16h a0 = gather_a(A, lane, 0);
  const v16h a1 = gather_a(A, lane, 1);
  const int roff = (lane >> 4) << 3;      // rows r / r+8 across lane halves

  // ---- semantic mask bits (D-tile lanes hold (row, class); assemble via ds_or)
  if (lane < 16) mwb[wave][lane] = 0u;
#pragma unroll
  for (int st = 0; st < 2; ++st) {
    v8f acc = {};
    acc = wmma16(a0, gather_b(WsT, lane, 0, st), acc);
    acc = wmma16(a1, gather_b(WsT, lane, 1, st), acc);
    const int cls = st * 16 + (lane & 15);
    if (cls < N_CLS) {
      const float bs = b_sem[cls];
#pragma unroll
      for (int r = 0; r < 8; ++r)
        if (acc[r] + bs > SEM_LOGIT) atomicOr(&mwb[wave][r + roff], 1u << cls);
    }
  }
  if (lane < 16) maskw[rows0 + lane] = mwb[wave][lane];  // same-wave LDS is in-order

  // ---- h1 = elu(A @ W1 * g1 + b1) -> Hsh
#pragma unroll
  for (int nt = 0; nt < 4; ++nt) {
    v8f acc = {};
    acc = wmma16(a0, gather_b(W1t, lane, 0, nt), acc);
    acc = wmma16(a1, gather_b(W1t, lane, 1, nt), acc);
    const int col = nt * 16 + (lane & 15);
    const float g = g_o1[col], b = b_o1[col];
#pragma unroll
    for (int r = 0; r < 8; ++r)
      H[(r + roff) * A_STRIDE + col] = (_Float16)elu1(acc[r] * g + b);
  }
  const v16h h0 = gather_a(H, lane, 0);
  const v16h h1 = gather_a(H, lane, 1);

  // ---- h2 = elu(h1 @ W2 * g2 + b2) -> reuse Ash (a0/a1 already in regs)
#pragma unroll
  for (int nt = 0; nt < 4; ++nt) {
    v8f acc = {};
    acc = wmma16(h0, gather_b(W2t, lane, 0, nt), acc);
    acc = wmma16(h1, gather_b(W2t, lane, 1, nt), acc);
    const int col = nt * 16 + (lane & 15);
    const float g = g_o2[col], b = b_o2[col];
#pragma unroll
    for (int r = 0; r < 8; ++r)
      A[(r + roff) * A_STRIDE + col] = (_Float16)elu1(acc[r] * g + b);
  }
  const v16h g0 = gather_a(A, lane, 0);
  const v16h g1 = gather_a(A, lane, 1);

  // ---- offset = h2 @ W_o3 (cols 0..2), then clipped vote
  v8f acc3 = {};
  acc3 = wmma16(g0, gather_b(W3t, lane, 0, 0), acc3);
  acc3 = wmma16(g1, gather_b(W3t, lane, 1, 0), acc3);
  const int j = lane & 15;
  if (j < 3) {
    const float minb = (float)(mm[j] - 1) * VOXEL;
    const float maxb = (float)(mm[4 + j] + 1) * VOXEL;
#pragma unroll
    for (int r = 0; r < 8; ++r) {
      const int n = rows0 + r + roff;
      const float x = (float)coords[n * 4 + 1 + j] * VOXEL;
      voted[(size_t)n * 3 + j] = fminf(fmaxf(x + acc3[r], minb), maxb);
    }
  }
}

// ---------------------------------------------------------------------------
// Kernel 3: per-class branch conv + BN + ELU + mask, fused heads.
// Block owns 128 rows; feats A-operands kept in registers while looping over
// all 18 classes (feats touches HBM exactly once). W_ci[c+1] is prefetched
// into LDS by the Tensor Data Mover while class c runs on the WMMA pipes.
// ---------------------------------------------------------------------------
__global__ __launch_bounds__(256) void per_class_head_kernel(
    const float* __restrict__ feats, const float* __restrict__ W_ci,
    const float* __restrict__ g_ci, const float* __restrict__ b_ci,
    const float* __restrict__ W_ctr, const float* __restrict__ W_reg,
    const float* __restrict__ W_cls, const float* __restrict__ b_cls,
    const float* __restrict__ scales, const unsigned* __restrict__ maskw,
    float* __restrict__ out) {
  __shared__ _Float16 Bt[64 * B_STRIDE];     // W_ci[c]^T (f16, B-operand source)
  __shared__ _Float16 Ht[16 * B_STRIDE];     // fused head [ctr | reg x6 | cls_c | 0...]^T
  __shared__ _Float16 Ash[8][16 * A_STRIDE]; // per-wave: feats tile, then feat_c tile
#if HAVE_TDM
  __shared__ float Wstage[2][4096];          // TDM double-buffer for W_ci[c] (fp32)
#endif

  const int tid = threadIdx.x, wave = tid >> 5, lane = tid & 31;
  const int rows0 = blockIdx.x * 128 + wave * 16;
  _Float16* A = Ash[wave];

  load_a_tile(feats + (size_t)rows0 * C_FEAT, A, lane);
  const v16h a0 = gather_a(A, lane, 0);
  const v16h a1 = gather_a(A, lane, 1);
  const int roff = (lane >> 4) << 3;

  unsigned mrow[8];
#pragma unroll
  for (int r = 0; r < 8; ++r) mrow[r] = maskw[rows0 + r + roff];

#if HAVE_TDM
  if (wave == 0) tdm_load_w64x64(W_ci, &Wstage[0][0]);   // prefetch class 0
#endif

  for (int c = 0; c < N_CLS; ++c) {
#if HAVE_TDM
    if (wave == 0) __builtin_amdgcn_s_wait_tensorcnt(0); // stage[c&1] landed in LDS
    __syncthreads();  // publish staged data; all waves done reading previous Bt/Ht
    const float* W = Wstage[c & 1];
#else
    __syncthreads();  // all waves done reading previous class's Bt/Ht
    const float* W = W_ci + (size_t)c * 4096;
#endif
    for (int i = tid; i < 4096; i += 256) {
      const int d = i >> 6, e = i & 63;
      Bt[e * B_STRIDE + d] = (_Float16)W[i];
    }
    for (int i = tid; i < 16 * 64; i += 256) {
      const int j = i >> 6, d = i & 63;
      float v = 0.0f;
      if (j == 0)      v = W_ctr[d];
      else if (j <= 6) v = W_reg[d * 6 + (j - 1)];
      else if (j == 7) v = W_cls[d * N_CLS + c];
      Ht[j * B_STRIDE + d] = (_Float16)v;
    }
#if HAVE_TDM
    if (wave == 0 && c + 1 < N_CLS)   // async prefetch of next class overlaps WMMA below
      tdm_load_w64x64(W_ci + (size_t)(c + 1) * 4096, &Wstage[(c + 1) & 1][0]);
#endif
    __syncthreads();

    // feat_c tile = elu(A @ W_ci[c] * g + b) * mask  -> Ash (f16)
#pragma unroll
    for (int nt = 0; nt < 4; ++nt) {
      v8f acc = {};
      acc = wmma16(a0, gather_b(Bt, lane, 0, nt), acc);
      acc = wmma16(a1, gather_b(Bt, lane, 1, nt), acc);
      const int col = nt * 16 + (lane & 15);
      const float g = g_ci[c * C_FEAT + col], b = b_ci[c * C_FEAT + col];
#pragma unroll
      for (int r = 0; r < 8; ++r) {
        float v = elu1(acc[r] * g + b);
        if (!((mrow[r] >> c) & 1u)) v = 0.0f;
        A[(r + roff) * A_STRIDE + col] = (_Float16)v;
      }
    }

    // heads: one fused 16x16 tile (cols: 0=ctr, 1..6=reg, 7=cls, 8..15 pad)
    v8f acc2 = {};
    acc2 = wmma16(gather_a(A, lane, 0), gather_b(Ht, lane, 0, 0), acc2);
    acc2 = wmma16(gather_a(A, lane, 1), gather_b(Ht, lane, 1, 0), acc2);
    const float sc = scales[c], bc = b_cls[c];
    const int j = lane & 15;
    if (j < 8) {
#pragma unroll
      for (int r = 0; r < 8; ++r) {
        const int n = rows0 + r + roff;
        float v = acc2[r];
        if (j >= 1 && j <= 6) v = __expf(v * sc);
        else if (j == 7)      v += bc;
        out[((size_t)c * N_PTS + n) * 8 + j] = v;
      }
    }
  }
}

// ---------------------------------------------------------------------------
extern "C" void kernel_launch(void* const* d_in, const int* in_sizes, int n_in,
                              void* d_out, int out_size, void* d_ws, size_t ws_size,
                              hipStream_t stream) {
  (void)in_sizes; (void)n_in; (void)out_size; (void)ws_size;
  const int*   coords = (const int*)d_in[0];
  const float* feats  = (const float*)d_in[1];
  const float* W_sem  = (const float*)d_in[2];
  const float* b_sem  = (const float*)d_in[3];
  const float* W_o1   = (const float*)d_in[4];
  const float* g_o1   = (const float*)d_in[5];
  const float* b_o1   = (const float*)d_in[6];
  const float* W_o2   = (const float*)d_in[7];
  const float* g_o2   = (const float*)d_in[8];
  const float* b_o2   = (const float*)d_in[9];
  const float* W_o3   = (const float*)d_in[10];
  const float* W_ci   = (const float*)d_in[11];
  const float* g_ci   = (const float*)d_in[12];
  const float* b_ci   = (const float*)d_in[13];
  const float* W_ctr  = (const float*)d_in[14];
  const float* W_reg  = (const float*)d_in[15];
  const float* W_cls  = (const float*)d_in[16];
  const float* b_cls  = (const float*)d_in[17];
  const float* scales = (const float*)d_in[18];

  int*      mm    = (int*)d_ws;                 // 8 ints: min xyz, pad, max xyz, pad
  unsigned* maskw = (unsigned*)d_ws + 8;        // N_PTS class-mask words

  float* out   = (float*)d_out;                            // [18, N, 8]
  float* voted = out + (size_t)N_CLS * N_PTS * 8;          // [N, 3]

  minmax_init<<<1, 32, 0, stream>>>(mm);
  minmax_kernel<<<512, 256, 0, stream>>>(coords, mm);
  offset_sem_vote_kernel<<<N_PTS / 128, 256, 0, stream>>>(
      coords, feats, W_sem, b_sem, W_o1, g_o1, b_o1, W_o2, g_o2, b_o2, W_o3,
      mm, maskw, voted);
  per_class_head_kernel<<<N_PTS / 128, 256, 0, stream>>>(
      feats, W_ci, g_ci, b_ci, W_ctr, W_reg, W_cls, b_cls, scales, maskw, out);
}
